// MLLA_EEG_Block_75385265979626
// MI455X (gfx1250) — matrix operand order repaired and checked by reading, not compiled, not run on hardware
//
#include <hip/hip_runtime.h>
#include <hip/hip_bf16.h>
#include <math.h>

// ---------------- problem constants ----------------
constexpr int Bb  = 128;
constexpr int Nn  = 200;
constexpr int Cc  = 1024;
constexpr int Hh  = 16;
constexpr int Dd  = 64;          // Cc / Hh
constexpr int HIDd = 4096;       // 4*Cc
constexpr long long BN = (long long)Bb * Nn;          // 25600
constexpr long long SZ = BN * Cc;                     // 26,214,400

typedef __attribute__((ext_vector_type(16))) __bf16 v16bf;
typedef __attribute__((ext_vector_type(8)))  float  v8f;

// ---------------- activation helper ----------------
__device__ __forceinline__ float apply_act(float x, int act) {
    if (act == 1) {                       // silu
        return x / (1.f + __expf(-x));
    } else if (act == 2) {                // elu(x)+1
        return (x > 0.f) ? (x + 1.f) : __expf(x);
    } else if (act == 3) {                // exact gelu
        return 0.5f * x * (1.f + erff(x * 0.70710678118f));
    }
    return x;
}

// =====================================================================
// WMMA GEMM: out[M,Nout] = act(A[M,K] @ W[K,Nout] + bias) (+ residual)
// A bf16 row-major [M][K]; WT bf16 *transposed* [Nout][K].
// Block 256 thr = 8 waves; block tile 128x128; wave tile 32x64 (2x4 WMMA).
// Double-buffered LDS filled with async global->LDS copies (ASYNCcnt).
// Requires M%128==0, Nout%128==0, K%32==0 (true for all uses here).
// =====================================================================
#define BM  128
#define BNB 128
#define BK  32

__global__ __launch_bounds__(256)
void wmma_gemm_kernel(const __bf16* __restrict__ A, const __bf16* __restrict__ WT,
                      const float* __restrict__ bias, const float* __restrict__ residual,
                      float* __restrict__ outF, __bf16* __restrict__ outB,
                      int M, int K, int Nout, int act)
{
    __shared__ __bf16 ldsA[2][BM * BK];     // [row][k]
    __shared__ __bf16 ldsB[2][BNB * BK];    // [n][k]

    const int tid  = threadIdx.x;
    const int wave = tid >> 5;
    const int lane = tid & 31;
    const int wm   = wave >> 1;             // 0..3 -> M offset wm*32
    const int wn   = wave & 1;              // 0..1 -> N offset wn*64
    const size_t bm = (size_t)blockIdx.x * BM;
    const size_t bn = (size_t)blockIdx.y * BNB;

    // staging geometry: thread copies rows sr, sr+64 (16B of K each)
    const int sr = tid >> 2;                // 0..63
    const int sc = (tid & 3) * 8;           // k element offset 0,8,16,24
    const __bf16* gA = A  + (bm + sr) * (size_t)K + sc;
    const __bf16* gB = WT + (bn + sr) * (size_t)K + sc;

    auto stage = [&](int buf, int k0) {
        #pragma unroll
        for (int i = 0; i < 2; ++i) {
            const unsigned la = (unsigned)(size_t)&ldsA[buf][(sr + i * 64) * BK + sc];
            const __bf16* pa = gA + (size_t)i * 64 * K + k0;
            asm volatile("global_load_async_to_lds_b128 %0, %1, off"
                         :: "v"(la), "v"(pa) : "memory");
            const unsigned lb = (unsigned)(size_t)&ldsB[buf][(sr + i * 64) * BK + sc];
            const __bf16* pb = gB + (size_t)i * 64 * K + k0;
            asm volatile("global_load_async_to_lds_b128 %0, %1, off"
                         :: "v"(lb), "v"(pb) : "memory");
        }
    };

    v8f acc[2][4] = {};

    auto compute = [&](int buf) {
        union Frag { v16bf v; uint4 u[2]; };
        Frag a[2], b[4];
        // A fragment: lanes 0-15 rows M=lane, K runs 0..7/16..23; lanes 16-31 K runs 8..15/24..31
        const int arow = wm * 32 + (lane & 15);
        const int akb  = (lane >= 16) ? 8 : 0;
        #pragma unroll
        for (int i = 0; i < 2; ++i) {
            const __bf16* p = &ldsA[buf][(arow + i * 16) * BK];
            a[i].u[0] = *(const uint4*)(p + akb);
            a[i].u[1] = *(const uint4*)(p + akb + 16);
        }
        // B fragment: lane holds column n = lane&15, K runs 0..7/16..23 (lanes<16) / 8..15/24..31
        const int bcol = wn * 64 + (lane & 15);
        const int bkb  = (lane >= 16) ? 8 : 0;
        #pragma unroll
        for (int j = 0; j < 4; ++j) {
            const __bf16* p = &ldsB[buf][(bcol + j * 16) * BK + bkb];
            b[j].u[0] = *(const uint4*)(p);
            b[j].u[1] = *(const uint4*)(p + 16);
        }
        #pragma unroll
        for (int i = 0; i < 2; ++i)
            #pragma unroll
            for (int j = 0; j < 4; ++j)
                acc[i][j] = __builtin_amdgcn_wmma_f32_16x16x32_bf16(
                    false, a[i].v, false, b[j].v, (short)0, acc[i][j], false, false);
    };

    const int nIter = K / BK;
    // prologue: stage tile 0
    stage(0, 0);
    asm volatile("s_wait_asynccnt 0x0" ::: "memory");
    __syncthreads();

    for (int it = 0; it < nIter - 1; ++it) {
        const int cur = it & 1;
        stage(cur ^ 1, (it + 1) * BK);                       // async fill of next buffer
        __builtin_prefetch(gA + (size_t)(it + 2) * BK, 0, 0); // speculative, branch-free
        __builtin_prefetch(gB + (size_t)(it + 2) * BK, 0, 0);
        compute(cur);                                        // overlaps with async DMA
        asm volatile("s_wait_asynccnt 0x0" ::: "memory");
        __syncthreads();
    }
    compute((nIter - 1) & 1);

    // ---- epilogue: D layout = row v + 8*(lane>=16), col lane&15 ----
    const int colBase = (int)bn + wn * 64 + (lane & 15);
    const int rowHi   = (lane >= 16) ? 8 : 0;
    #pragma unroll
    for (int i = 0; i < 2; ++i) {
        #pragma unroll
        for (int j = 0; j < 4; ++j) {
            const int col = colBase + j * 16;
            const float bv = bias ? bias[col] : 0.f;
            #pragma unroll
            for (int v = 0; v < 8; ++v) {
                const int row = (int)bm + wm * 32 + i * 16 + rowHi + v;
                float x = acc[i][j][v] + bv;
                x = apply_act(x, act);
                const size_t idx = (size_t)row * Nout + col;
                if (residual) x += residual[idx];
                if (outF) outF[idx] = x;
                if (outB) outB[idx] = (__bf16)x;
            }
        }
    }
}

// =====================================================================
// depthwise conv1d (k=3, pad 1 along N), fused residual / silu / bf16 out
// =====================================================================
__global__ void dwconv_kernel(const float* __restrict__ src, const float* __restrict__ w3,
                              const float* __restrict__ bias, const float* __restrict__ base,
                              float* __restrict__ outF, __bf16* __restrict__ outB, int applySilu)
{
    const size_t i = (size_t)blockIdx.x * blockDim.x + threadIdx.x;
    if (i >= (size_t)SZ) return;
    const int c = (int)(i % Cc);
    const size_t bn = i / Cc;
    const int n = (int)(bn % Nn);
    float v = bias[c] + w3[c * 3 + 1] * src[i];
    if (n > 0)      v += w3[c * 3 + 0] * src[i - Cc];
    if (n < Nn - 1) v += w3[c * 3 + 2] * src[i + Cc];
    if (applySilu) v = v / (1.f + __expf(-v));
    if (base) v += base[i];
    if (outF) outF[i] = v;
    if (outB) outB[i] = (__bf16)v;
}

// =====================================================================
// LayerNorm over C, bf16 output
// =====================================================================
__global__ __launch_bounds__(256)
void ln_kernel(const float* __restrict__ x, const float* __restrict__ g,
               const float* __restrict__ b, __bf16* __restrict__ out)
{
    __shared__ float red[256];
    const int row = blockIdx.x;
    const float* xr = x + (size_t)row * Cc;
    float s = 0.f;
    for (int c = threadIdx.x; c < Cc; c += 256) s += xr[c];
    red[threadIdx.x] = s; __syncthreads();
    for (int o = 128; o > 0; o >>= 1) {
        if (threadIdx.x < o) red[threadIdx.x] += red[threadIdx.x + o];
        __syncthreads();
    }
    const float mean = red[0] / Cc;
    __syncthreads();
    float vs = 0.f;
    for (int c = threadIdx.x; c < Cc; c += 256) { float d = xr[c] - mean; vs += d * d; }
    red[threadIdx.x] = vs; __syncthreads();
    for (int o = 128; o > 0; o >>= 1) {
        if (threadIdx.x < o) red[threadIdx.x] += red[threadIdx.x + o];
        __syncthreads();
    }
    const float rstd = rsqrtf(red[0] / Cc + 1e-5f);
    __bf16* orow = out + (size_t)row * Cc;
    for (int c = threadIdx.x; c < Cc; c += 256)
        orow[c] = (__bf16)((xr[c] - mean) * rstd * g[c] + b[c]);
}

// ---------------- k-mean over n, per (b,h,d) ----------------
__global__ void kmean_kernel(const float* __restrict__ qk, float* __restrict__ kmean)
{
    const int bh = blockIdx.x;
    const int b = bh / Hh, h = bh % Hh;
    const int d = threadIdx.x;                // 0..63
    const float* kp = qk + (size_t)b * Nn * (2 * Cc) + Cc + h * Dd + d;
    float s = 0.f;
    for (int n = 0; n < Nn; ++n) s += kp[(size_t)n * 2 * Cc];
    kmean[(size_t)bh * Dd + d] = s / Nn;
}

// ---------------- z = 1/(q . kmean + eps), per (b,h,n) ----------------
__global__ void z_kernel(const float* __restrict__ qk, const float* __restrict__ kmean,
                         float* __restrict__ z)
{
    const size_t i = (size_t)blockIdx.x * blockDim.x + threadIdx.x;
    const size_t total = (size_t)Bb * Hh * Nn;
    if (i >= total) return;
    const int n = (int)(i % Nn);
    const int bh = (int)(i / Nn);
    const int b = bh / Hh, h = bh % Hh;
    const float* qp = qk + ((size_t)b * Nn + n) * (2 * Cc) + h * Dd;
    const float* km = kmean + (size_t)bh * Dd;
    float s = 0.f;
    #pragma unroll 8
    for (int d = 0; d < Dd; ++d) s += qp[d] * km[d];
    z[i] = 1.f / (s + 1e-6f);
}

// ---------------- RoPE in place on q and k halves of qk ----------------
__global__ __launch_bounds__(256)
void rope_kernel(float* __restrict__ qk)
{
    const size_t row = blockIdx.x;            // b*N + n
    const int n = (int)(row % Nn);
    float* base = qk + row * (size_t)(2 * Cc);
    for (int p = threadIdx.x; p < 2 * (Cc / 2); p += 256) {  // 1024 pairs (q then k)
        const int which = p >> 9;             // 0=q, 1=k
        const int pp = p & 511;
        const int h = pp >> 5;
        const int j = pp & 31;                // pair index within head
        const int off = which * Cc + h * Dd + 2 * j;
        const float inv = __powf(10000.f, -(float)(2 * j) / (float)Dd);
        const float ang = (float)n * inv;
        float sn, cs;
        __sincosf(ang, &sn, &cs);
        const float x0 = base[off], x1 = base[off + 1];
        base[off]     = x0 * cs - x1 * sn;
        base[off + 1] = x0 * sn + x1 * cs;
    }
}

// ---------------- kv[b,h,d,e] = sum_n kr*v / N ----------------
__global__ __launch_bounds__(256)
void kv_kernel(const float* __restrict__ qk, const float* __restrict__ h2,
               float* __restrict__ kv)
{
    __shared__ float krow[Dd], vrow[Dd];
    const int bh = blockIdx.x;
    const int b = bh / Hh, h = bh % Hh;
    const int t = threadIdx.x;
    const int d  = t & 63;
    const int eg = t >> 6;                    // 0..3
    float acc[16] = {};
    for (int n = 0; n < Nn; ++n) {
        __syncthreads();
        if (t < 64)       krow[t]      = qk[((size_t)b * Nn + n) * (2 * Cc) + Cc + h * Dd + t];
        else if (t < 128) vrow[t - 64] = h2[((size_t)b * Nn + n) * Cc + h * Dd + (t - 64)];
        __syncthreads();
        const float kd = krow[d];
        #pragma unroll
        for (int i = 0; i < 16; ++i) acc[i] += kd * vrow[eg * 16 + i];
    }
    const float invN = 1.f / (float)Nn;
    #pragma unroll
    for (int i = 0; i < 16; ++i)
        kv[((size_t)bh * Dd + d) * Dd + eg * 16 + i] = acc[i] * invN;
}

// ---------------- o[b,n,h,e] = z * sum_d qr*kv ----------------
__global__ __launch_bounds__(256)
void o_kernel(const float* __restrict__ qk, const float* __restrict__ kvm,
              const float* __restrict__ z, float* __restrict__ o)
{
    __shared__ float kvs[Dd * Dd];
    __shared__ float qrows[4 * Dd];
    const int bh = blockIdx.x;
    const int b = bh / Hh, h = bh % Hh;
    const int t = threadIdx.x;
    for (int i = t; i < Dd * Dd; i += 256) kvs[i] = kvm[(size_t)bh * Dd * Dd + i];
    const int nn = t >> 6, e = t & 63;
    for (int n0 = 0; n0 < Nn; n0 += 4) {
        __syncthreads();
        {
            const int n = n0 + (t >> 6);
            if (n < Nn) qrows[t] = qk[((size_t)b * Nn + n) * (2 * Cc) + h * Dd + (t & 63)];
        }
        __syncthreads();
        const int n = n0 + nn;
        if (n < Nn) {
            float s = 0.f;
            #pragma unroll 8
            for (int d = 0; d < Dd; ++d) s += qrows[nn * 64 + d] * kvs[d * 64 + e];
            o[((size_t)b * Nn + n) * Cc + h * Dd + e] = s * z[(size_t)bh * Nn + n];
        }
    }
}

// ---------------- elementwise mul + bf16 cast ----------------
__global__ void mulcast_kernel(const float* __restrict__ a, const float* __restrict__ b,
                               __bf16* __restrict__ out, size_t n)
{
    const size_t i = (size_t)blockIdx.x * blockDim.x + threadIdx.x;
    if (i < n) out[i] = (__bf16)(a[i] * b[i]);
}

// ---------------- f32 W[K][N] -> bf16 WT[N][K] (LDS-tiled transpose) ----------------
__global__ __launch_bounds__(256)
void castT_kernel(const float* __restrict__ in, __bf16* __restrict__ out, int K, int Nout)
{
    __shared__ float tile[32][33];
    const int kb = blockIdx.x * 32, nb = blockIdx.y * 32;
    const int tx = threadIdx.x & 31, ty = threadIdx.x >> 5;   // 32 x 8
    #pragma unroll
    for (int i = ty; i < 32; i += 8)
        tile[i][tx] = in[(size_t)(kb + i) * Nout + nb + tx];
    __syncthreads();
    #pragma unroll
    for (int i = ty; i < 32; i += 8)
        out[(size_t)(nb + i) * K + kb + tx] = (__bf16)tile[tx][i];
}

// =====================================================================
// launcher
// =====================================================================
extern "C" void kernel_launch(void* const* d_in, const int* in_sizes, int n_in,
                              void* d_out, int out_size, void* d_ws, size_t ws_size,
                              hipStream_t stream)
{
    const float* x      = (const float*)d_in[0];
    const float* cpe1_w = (const float*)d_in[1];
    const float* cpe1_b = (const float*)d_in[2];
    const float* cpe2_w = (const float*)d_in[3];
    const float* cpe2_b = (const float*)d_in[4];
    const float* n1_g   = (const float*)d_in[5];
    const float* n1_b   = (const float*)d_in[6];
    const float* n2_g   = (const float*)d_in[7];
    const float* n2_b   = (const float*)d_in[8];
    const float* in_w   = (const float*)d_in[9];
    const float* in_b   = (const float*)d_in[10];
    const float* out_w  = (const float*)d_in[11];
    const float* out_b  = (const float*)d_in[12];
    const float* actp_w = (const float*)d_in[13];
    const float* actp_b = (const float*)d_in[14];
    const float* dwc_w  = (const float*)d_in[15];
    const float* dwc_b  = (const float*)d_in[16];
    const float* qk_w   = (const float*)d_in[17];
    const float* qk_b   = (const float*)d_in[18];
    const float* lepe_w = (const float*)d_in[19];
    const float* lepe_b = (const float*)d_in[20];
    const float* fc1_w  = (const float*)d_in[21];
    const float* fc1_b  = (const float*)d_in[22];
    const float* fc2_w  = (const float*)d_in[23];
    const float* fc2_b  = (const float*)d_in[24];
    float* outp = (float*)d_out;

    // ---------- workspace carve-up ----------
    char* base = (char*)d_ws;
    size_t off = 0;
    auto alloc = [&](size_t bytes) -> char* {
        char* p = base + off;
        off += (bytes + 255) & ~(size_t)255;
        return p;
    };
    float*  x0     = (float*)alloc(SZ * 4);            // x + cpe1 (shortcut)
    float*  actres = (float*)alloc(SZ * 4);
    float*  hbuf   = (float*)alloc(SZ * 4);            // h, later o
    float*  h2     = (float*)alloc(SZ * 4);
    float*  qkbuf  = (float*)alloc(2 * SZ * 4);        // later x1 (lo) / x2 (hi)
    __bf16* xnb    = (__bf16*)alloc(SZ * 2);           // later gated-output bf16
    __bf16* h2b    = (__bf16*)alloc(SZ * 2);           // later ln2 bf16
    __bf16* hidb   = (__bf16*)alloc(BN * HIDd * 2);
    float*  kmean  = (float*)alloc((size_t)Bb * Hh * Dd * 4);
    float*  zbuf   = (float*)alloc((size_t)Bb * Hh * Nn * 4);
    float*  kvbuf  = (float*)alloc((size_t)Bb * Hh * Dd * Dd * 4);
    __bf16* wactp  = (__bf16*)alloc((size_t)Cc * Cc * 2);      // all transposed [N][K]
    __bf16* win    = (__bf16*)alloc((size_t)Cc * Cc * 2);
    __bf16* wqk    = (__bf16*)alloc((size_t)Cc * 2 * Cc * 2);
    __bf16* wout   = (__bf16*)alloc((size_t)Cc * Cc * 2);
    __bf16* wfc1   = (__bf16*)alloc((size_t)Cc * HIDd * 2);
    __bf16* wfc2   = (__bf16*)alloc((size_t)HIDd * Cc * 2);
    float*  x1 = qkbuf;
    float*  x2 = qkbuf + SZ;

    const int TB = 256;
    auto blocks = [&](size_t n) { return (unsigned)((n + TB - 1) / TB); };

    // ---------- weight transpose-casts f32[K][N] -> bf16[N][K] ----------
    castT_kernel<<<dim3(Cc / 32, Cc / 32),   256, 0, stream>>>(actp_w, wactp, Cc, Cc);
    castT_kernel<<<dim3(Cc / 32, Cc / 32),   256, 0, stream>>>(in_w,   win,   Cc, Cc);
    castT_kernel<<<dim3(Cc / 32, 2*Cc / 32), 256, 0, stream>>>(qk_w,   wqk,   Cc, 2 * Cc);
    castT_kernel<<<dim3(Cc / 32, Cc / 32),   256, 0, stream>>>(out_w,  wout,  Cc, Cc);
    castT_kernel<<<dim3(Cc / 32, HIDd / 32), 256, 0, stream>>>(fc1_w,  wfc1,  Cc, HIDd);
    castT_kernel<<<dim3(HIDd / 32, Cc / 32), 256, 0, stream>>>(fc2_w,  wfc2,  HIDd, Cc);

    // ---------- 1. x0 = x + dwconv(x, cpe1) ----------
    dwconv_kernel<<<blocks(SZ), TB, 0, stream>>>(x, cpe1_w, cpe1_b, x, x0, nullptr, 0);

    // ---------- 2. xn = LN(x0) (bf16) ----------
    ln_kernel<<<(unsigned)BN, 256, 0, stream>>>(x0, n1_g, n1_b, xnb);

    // ---------- 3. act_res = silu(xn @ actp_w + b) ----------
    wmma_gemm_kernel<<<dim3((unsigned)(BN / BM), Cc / BNB), 256, 0, stream>>>(
        xnb, wactp, actp_b, nullptr, actres, nullptr, (int)BN, Cc, Cc, 1);

    // ---------- 4. h = xn @ in_w + b ----------
    wmma_gemm_kernel<<<dim3((unsigned)(BN / BM), Cc / BNB), 256, 0, stream>>>(
        xnb, win, in_b, nullptr, hbuf, nullptr, (int)BN, Cc, Cc, 0);

    // ---------- 5. h2 = silu(dwconv(h, dwc)) (f32 + bf16) ----------
    dwconv_kernel<<<blocks(SZ), TB, 0, stream>>>(hbuf, dwc_w, dwc_b, nullptr, h2, h2b, 1);

    // ---------- 6. qk = elu(h2 @ qk_w + b)+1 ----------
    wmma_gemm_kernel<<<dim3((unsigned)(BN / BM), 2 * Cc / BNB), 256, 0, stream>>>(
        h2b, wqk, qk_b, nullptr, qkbuf, nullptr, (int)BN, Cc, 2 * Cc, 2);

    // ---------- 7. kmean, z (pre-RoPE) ----------
    kmean_kernel<<<Bb * Hh, Dd, 0, stream>>>(qkbuf, kmean);
    z_kernel<<<blocks((size_t)Bb * Hh * Nn), TB, 0, stream>>>(qkbuf, kmean, zbuf);

    // ---------- 8. RoPE in place ----------
    rope_kernel<<<(unsigned)BN, 256, 0, stream>>>(qkbuf);

    // ---------- 9. kv = kr^T v / N ; o = qr kv * z (into hbuf) ----------
    kv_kernel<<<Bb * Hh, 256, 0, stream>>>(qkbuf, h2, kvbuf);
    o_kernel<<<Bb * Hh, 256, 0, stream>>>(qkbuf, kvbuf, zbuf, hbuf);

    // ---------- 10. o += dwconv(h2, lepe) ----------
    dwconv_kernel<<<blocks(SZ), TB, 0, stream>>>(h2, lepe_w, lepe_b, hbuf, hbuf, nullptr, 0);

    // ---------- 11. gated = (o * act_res) bf16 ----------
    mulcast_kernel<<<blocks(SZ), TB, 0, stream>>>(hbuf, actres, xnb, (size_t)SZ);

    // ---------- 12. x1 = gated @ out_w + b + shortcut ----------
    wmma_gemm_kernel<<<dim3((unsigned)(BN / BM), Cc / BNB), 256, 0, stream>>>(
        xnb, wout, out_b, x0, x1, nullptr, (int)BN, Cc, Cc, 0);

    // ---------- 13. x2 = x1 + dwconv(x1, cpe2) ----------
    dwconv_kernel<<<blocks(SZ), TB, 0, stream>>>(x1, cpe2_w, cpe2_b, x1, x2, nullptr, 0);

    // ---------- 14. y = LN(x2) (bf16) ----------
    ln_kernel<<<(unsigned)BN, 256, 0, stream>>>(x2, n2_g, n2_b, h2b);

    // ---------- 15. hid = gelu(y @ fc1_w + b) (bf16) ----------
    wmma_gemm_kernel<<<dim3((unsigned)(BN / BM), HIDd / BNB), 256, 0, stream>>>(
        h2b, wfc1, fc1_b, nullptr, nullptr, hidb, (int)BN, Cc, HIDd, 3);

    // ---------- 16. out = hid @ fc2_w + b + x2 ----------
    wmma_gemm_kernel<<<dim3((unsigned)(BN / BM), Cc / BNB), 256, 0, stream>>>(
        hidb, wfc2, fc2_b, x2, outp, nullptr, (int)BN, HIDd, Cc, 0);
}
